// MultiHeadAttention_51058571215081
// MI455X (gfx1250) — matrix operand order repaired and checked by reading, not compiled, not run on hardware
//
#include <hip/hip_runtime.h>
#include <stdint.h>

// Problem constants (match reference)
constexpr int kB  = 4;
constexpr int kS  = 2048;
constexpr int kD  = 1024;
constexpr int kH  = 16;
constexpr int kDK = 64;
constexpr int kM  = kB * kS;          // 8192 rows for projections

typedef __attribute__((ext_vector_type(16))) __bf16   v16bf;
typedef __attribute__((ext_vector_type(8)))  float    v8f;
typedef __attribute__((ext_vector_type(4)))  uint32_t u32x4;
typedef __attribute__((ext_vector_type(4)))  float    f32x4;
typedef __attribute__((ext_vector_type(4)))  int      i32x4;
typedef __attribute__((ext_vector_type(8)))  int      i32x8;

#if __has_builtin(__builtin_amdgcn_tensor_load_to_lds) && \
    __has_builtin(__builtin_amdgcn_s_wait_tensorcnt)
#define USE_TDM 1
#else
#define USE_TDM 0
#endif

union BF16x16 {
  v16bf    v;
  u32x4    q[2];
  uint32_t u[8];
  uint16_t h[16];
};

__device__ inline uint16_t f2bf(float f) {
  uint32_t u = __float_as_uint(f);
  uint32_t r = u + 0x7FFFu + ((u >> 16) & 1u);   // round-to-nearest-even
  return (uint16_t)(r >> 16);
}

__device__ inline v8f wmma_bf16(v16bf a, v16bf b, v8f c) {
  return __builtin_amdgcn_wmma_f32_16x16x32_bf16(
      /*neg_a=*/false, a, /*neg_b=*/false, b,
      /*c_mod=*/(short)0, c, /*reuse_a=*/false, /*reuse_b=*/false);
}

// A 16x32 bf16 fragment: lane=row; lanes 0-15: K {0..7,16..23}, lanes 16-31:
// K {8..15,24..31}.  Both K-halves are 8 contiguous bf16 -> two b128 loads.
__device__ inline void load_frag_a(BF16x16& f, const uint16_t* row, int half) {
  f.q[0] = *(const u32x4*)(row + 8 * half);
  f.q[1] = *(const u32x4*)(row + 16 + 8 * half);
}
// B 32x16 bf16 fragment: lane=col; lanes 0-15: K {0..7,16..23}, lanes 16-31:
// K {8..15,24..31} (operand K-major in memory) -> two b128 loads.
__device__ inline void load_frag_b(BF16x16& f, const uint16_t* row, int half) {
  f.q[0] = *(const u32x4*)(row + 8 * half);
  f.q[1] = *(const u32x4*)(row + 16 + 8 * half);
}

// ds_swizzle XOR shuffle, group-of-32 encoding {0,xor,or=0,and=0x1F}
template <int XORM>
__device__ inline float swz_xor(float v) {
  int i = __builtin_amdgcn_ds_swizzle(__float_as_int(v), (XORM << 10) | 0x1F);
  return __int_as_float(i);
}
__device__ inline float rmax16(float v) {
  v = fmaxf(v, swz_xor<1>(v));
  v = fmaxf(v, swz_xor<2>(v));
  v = fmaxf(v, swz_xor<4>(v));
  v = fmaxf(v, swz_xor<8>(v));
  return v;
}
__device__ inline float rsum16(float v) {
  v += swz_xor<1>(v);
  v += swz_xor<2>(v);
  v += swz_xor<4>(v);
  v += swz_xor<8>(v);
  return v;
}

#if USE_TDM
// Issue a TDM 2-D tile load (bf16 elements) into LDS.  D# bitfields per
// CDNA5 ISA: group0 = {count=1, lds_addr, global_addr,
// type=2}; group1 = {data_size=2B, tensor dims, tile dims, dim0 stride}.
__device__ inline void tdm_load_2d(uint32_t lds_byte_addr, const void* gptr,
                                   uint32_t tile_d0, uint32_t tile_d1,
                                   uint32_t tensor_d0, uint32_t tensor_d1,
                                   uint32_t stride0_elems) {
  uint64_t ga = (uint64_t)(uintptr_t)gptr;
  u32x4 g0;
  g0[0] = 1u;                                        // count=1 (valid), user D#
  g0[1] = lds_byte_addr;                             // lds_addr [63:32]
  g0[2] = (uint32_t)ga;                              // global_addr [95:64]
  g0[3] = (uint32_t)((ga >> 32) & 0x01FFFFFFu) | (2u << 30);  // addr hi + type=2
  i32x8 g1;
  g1[0] = (int)(1u << 16);                           // data_size=1 (2 bytes)
  g1[1] = (int)((tensor_d0 & 0xFFFFu) << 16);        // tensor_dim0[15:0]
  g1[2] = (int)(((tensor_d0 >> 16) & 0xFFFFu) | ((tensor_d1 & 0xFFFFu) << 16));
  g1[3] = (int)(((tensor_d1 >> 16) & 0xFFFFu) | ((tile_d0 & 0xFFFFu) << 16));
  g1[4] = (int)(tile_d1 & 0xFFFFu);                  // tile_dim1, tile_dim2=0
  g1[5] = (int)stride0_elems;                        // tensor_dim0_stride lo32
  g1[6] = 0;
  g1[7] = 0;
  i32x4 z = {0, 0, 0, 0};
#if __clang_major__ >= 23
  i32x8 z8 = {0, 0, 0, 0, 0, 0, 0, 0};
  __builtin_amdgcn_tensor_load_to_lds(g0, g1, z, z, z8, 0);
#else
  __builtin_amdgcn_tensor_load_to_lds(g0, g1, z, z, 0);
#endif
}
#endif  // USE_TDM

// ---------------------------------------------------------------------------
// Kernel 1a: fp32 -> bf16, vectorized (8 elems / thread / step)
// ---------------------------------------------------------------------------
__global__ __launch_bounds__(256) void k_cvt(const float* __restrict__ in,
                                             uint16_t* __restrict__ out,
                                             int n8) {
  for (int i = blockIdx.x * blockDim.x + threadIdx.x; i < n8;
       i += gridDim.x * blockDim.x) {
    f32x4 a = ((const f32x4*)in)[2 * i];
    f32x4 b = ((const f32x4*)in)[2 * i + 1];
    union { u32x4 q; uint16_t h[8]; } o;
#pragma unroll
    for (int j = 0; j < 4; ++j) { o.h[j] = f2bf(a[j]); o.h[4 + j] = f2bf(b[j]); }
    ((u32x4*)out)[i] = o.q;
  }
}

// ---------------------------------------------------------------------------
// Kernel 1b: fp32 -> bf16 with transpose (W[K][N] -> Wt[N][K]), LDS-tiled.
// ---------------------------------------------------------------------------
__global__ __launch_bounds__(256) void k_cvt_w_t(const float* __restrict__ W,
                                                 uint16_t* __restrict__ Wt) {
  __shared__ uint16_t tile[32][33];
  const int tx = threadIdx.x;            // 0..31
  const int ty = threadIdx.y;            // 0..7
  const int n0 = blockIdx.x * 32;
  const int k0 = blockIdx.y * 32;
#pragma unroll
  for (int j = 0; j < 4; ++j)
    tile[ty * 4 + j][tx] = f2bf(W[(size_t)(k0 + ty * 4 + j) * kD + n0 + tx]);
  __syncthreads();
#pragma unroll
  for (int j = 0; j < 4; ++j)
    Wt[(size_t)(n0 + ty * 4 + j) * kD + k0 + tx] = tile[tx][ty * 4 + j];
}

// ---------------------------------------------------------------------------
// GEMM helpers: software-pipelined fragment sets (double-buffered registers)
// ---------------------------------------------------------------------------
struct Frags {
  BF16x16 a;
  BF16x16 b[4];
};
__device__ inline void load_set(Frags& f, const uint16_t* arow,
                                const uint16_t* brow0, int k0, int half) {
  load_frag_a(f.a, arow + k0, half);
#pragma unroll
  for (int t = 0; t < 4; ++t)
    load_frag_b(f.b[t], brow0 + (size_t)(16 * t) * kD + k0, half);
}
__device__ inline void mma_set(v8f acc[4], const Frags& f) {
#pragma unroll
  for (int t = 0; t < 4; ++t) acc[t] = wmma_bf16(f.a.v, f.b[t].v, acc[t]);
}

// ---------------------------------------------------------------------------
// Kernel 2: bf16 GEMM  C[M x D] = X[M x D] * Wt^T (Wt is N-major).
// One wave computes a 16x64 C strip, K-loop software pipelined.
// vmode==0: head-split bf16 Out[B,H,S,DK];  vmode==1: transposed Vt[B,H,DK,S].
// ---------------------------------------------------------------------------
__global__ __launch_bounds__(256) void k_proj(const uint16_t* __restrict__ X,
                                              const uint16_t* __restrict__ Wt,
                                              uint16_t* __restrict__ Out,
                                              int vmode) {
  const int lane = threadIdx.x & 31;
  const int wave = threadIdx.x >> 5;
  const int half = lane >> 4;
  const int l15  = lane & 15;
  const int m0 = blockIdx.x * 16;
  const int n0 = blockIdx.y * 512 + wave * 64;

  v8f acc[4] = {};
  const uint16_t* arow  = X  + (size_t)(m0 + l15) * kD;
  const uint16_t* brow0 = Wt + (size_t)(n0 + l15) * kD;

  Frags fA, fB;
  load_set(fA, arow, brow0, 0, half);
  for (int k0 = 0; k0 < kD; k0 += 64) {
    load_set(fB, arow, brow0, k0 + 32, half);   // prefetch next set
    mma_set(acc, fA);
    if (k0 + 64 < kD) load_set(fA, arow, brow0, k0 + 64, half);
    mma_set(acc, fB);
  }

  // C layout: lane=col(n), VGPR r = row r + 8*half.
  const int bidx = m0 / kS;
  const int srow = m0 % kS;
  const int h    = n0 / kDK;
  const size_t bh = (size_t)bidx * kH + h;
#pragma unroll
  for (int t = 0; t < 4; ++t)
#pragma unroll
    for (int r = 0; r < 8; ++r) {
      int s  = srow + r + 8 * half;
      int dk = 16 * t + l15;
      if (vmode)
        Out[(bh * kDK + dk) * kS + s] = f2bf(acc[t][r]);   // Vt[b,h,dk,s]
      else
        Out[(bh * kS + s) * kDK + dk] = f2bf(acc[t][r]);   // Q/K[b,h,s,dk]
    }
}

// ---------------------------------------------------------------------------
// Kernel 3: flash-style causal attention, one wave per (b,h,16-query tile).
// 4 waves / block; K and Vt chunks staged in LDS by the Tensor Data Mover
// (fallback: direct b128 global loads).  P transpose goes through LDS.
// ---------------------------------------------------------------------------
__global__ __launch_bounds__(128) void k_attention(const uint16_t* __restrict__ Q,
                                                   const uint16_t* __restrict__ K,
                                                   const uint16_t* __restrict__ Vt,
                                                   uint16_t* __restrict__ O) {
  // per-wave LDS: K tile 32x64 (2048), Vt tile 64x32 (2048), P tile 16x32 (512)
  constexpr int kWaveLds = 2048 + 2048 + 512;          // elements
  __shared__ __align__(16) uint16_t smem[4 * kWaveLds];

  const int lane = threadIdx.x & 31;
  const int wave = threadIdx.x >> 5;
  const int half = lane >> 4;
  const int l15  = lane & 15;

  const int wid   = blockIdx.x * 4 + wave;       // 8192 waves total
  const int bh    = wid >> 7;
  const int qt    = wid & 127;
  const int qbase = qt * 16;
  const size_t base = (size_t)bh * kS * kDK;

  uint16_t* ldsK = smem + wave * kWaveLds;
  uint16_t* ldsV = ldsK + 2048;
  uint16_t* ldsP = ldsV + 2048;
#if USE_TDM
  const uint32_t ldsK_b = (uint32_t)(wave * kWaveLds) * 2u;
  const uint32_t ldsV_b = ldsK_b + 4096u;
#endif

  // Q A-fragments for dk 0-31 / 32-63 (b128 loads)
  BF16x16 aq[2];
  const uint16_t* qrow = Q + base + (size_t)(qbase + l15) * kDK;
#pragma unroll
  for (int t = 0; t < 2; ++t) {
    aq[t].q[0] = *(const u32x4*)(qrow + 32 * t + 8 * half);
    aq[t].q[1] = *(const u32x4*)(qrow + 32 * t + 16 + 8 * half);
  }

  float mrow[8], lrow[8];
  v8f acc[4] = {};
#pragma unroll
  for (int r = 0; r < 8; ++r) { mrow[r] = -INFINITY; lrow[r] = 0.0f; }

  const float scale = 0.125f;                    // 1/sqrt(DK)
  const int nch = (qbase + 15) / 32 + 1;         // causal 32-key chunks

  for (int c = 0; c < nch; ++c) {
    const int kbase = c * 32;

#if USE_TDM
    // prior chunk's LDS reads must be in VGPRs before TDM overwrites the tiles
    asm volatile("s_wait_dscnt 0" ::: "memory");
    // K chunk: 32 rows x 64 contiguous bf16 (row stride 64)
    tdm_load_2d(ldsK_b, K + base + (size_t)kbase * kDK,
                /*tile*/ 64, 32, /*tensor*/ 64, kS, /*stride0*/ 64);
    // Vt chunk: 64 rows x 32 contiguous bf16 (row stride kS)
    tdm_load_2d(ldsV_b, Vt + base + (size_t)kbase,
                /*tile*/ 32, 64, /*tensor*/ kS, 64, /*stride0*/ kS);
    __builtin_amdgcn_s_wait_tensorcnt((short)0);
#endif

    // ---- K fragments + score WMMAs -------------------------------------
    v8f s[2] = {};
#pragma unroll
    for (int st = 0; st < 2; ++st) {
#if USE_TDM
      const uint16_t* kp = ldsK + (16 * st + l15) * 64;
#else
      const uint16_t* kp = K + base + (size_t)(kbase + 16 * st + l15) * kDK;
#endif
#pragma unroll
      for (int kk = 0; kk < 2; ++kk) {
        BF16x16 bk;
        load_frag_b(bk, kp + 32 * kk, half);
        s[st] = wmma_bf16(aq[kk].v, bk.v, s[st]);
      }
    }

    // ---- V fragments loaded early so they overlap the softmax VALU work -
    BF16x16 bv[4];
#pragma unroll
    for (int t = 0; t < 4; ++t) {
#if USE_TDM
      const uint16_t* vp = ldsV + (16 * t + l15) * 32;
#else
      const uint16_t* vp = Vt + base + (size_t)(16 * t + l15) * kS + kbase;
#endif
      load_frag_b(bv[t], vp, half);
    }

    // ---- causal mask + online softmax (lane=key col, VGPR r = row) ------
    float corr[8];
#pragma unroll
    for (int r = 0; r < 8; ++r) {
      const int q    = qbase + r + 8 * half;
      const int col0 = kbase + l15;
      const int col1 = col0 + 16;
      float x0 = (col0 <= q) ? s[0][r] * scale : -INFINITY;
      float x1 = (col1 <= q) ? s[1][r] * scale : -INFINITY;
      float mx = rmax16(fmaxf(x0, x1));
      float mn = fmaxf(mrow[r], mx);
      corr[r]  = (mrow[r] == -INFINITY) ? 0.0f : __expf(mrow[r] - mn);
      float p0 = __expf(x0 - mn);
      float p1 = __expf(x1 - mn);
      float rs = rsum16(p0 + p1);
      lrow[r]  = lrow[r] * corr[r] + rs;
      mrow[r]  = mn;
      ldsP[(r + 8 * half) * 32 + l15]      = f2bf(p0);
      ldsP[(r + 8 * half) * 32 + l15 + 16] = f2bf(p1);
    }
#pragma unroll
    for (int t = 0; t < 4; ++t)
#pragma unroll
      for (int r = 0; r < 8; ++r) acc[t][r] *= corr[r];

    asm volatile("s_wait_dscnt 0" ::: "memory");

    // reload P as A-fragment [16 queries x 32 keys] via ds b128 loads
    BF16x16 ap;
    ap.q[0] = *(const u32x4*)(&ldsP[l15 * 32 + 8 * half]);
    ap.q[1] = *(const u32x4*)(&ldsP[l15 * 32 + 16 + 8 * half]);

    // ---- acc[16 x 64] += P[16x32] * V[32x64] ----------------------------
#pragma unroll
    for (int t = 0; t < 4; ++t) acc[t] = wmma_bf16(ap.v, bv[t].v, acc[t]);
  }

  // epilogue: divide by l, merge heads into [B, S, D] bf16
  const int b_ = bh >> 4;
  const int h_ = bh & 15;
  float inv[8];
#pragma unroll
  for (int r = 0; r < 8; ++r) inv[r] = (lrow[r] > 0.0f) ? 1.0f / lrow[r] : 0.0f;
#pragma unroll
  for (int t = 0; t < 4; ++t)
#pragma unroll
    for (int r = 0; r < 8; ++r) {
      int row = qbase + r + 8 * half;
      O[((size_t)b_ * kS + row) * kD + h_ * kDK + 16 * t + l15] =
          f2bf(acc[t][r] * inv[r]);
    }
}

// ---------------------------------------------------------------------------
// Kernel 4: output projection, software-pipelined bf16 GEMM, fp32 store
// ---------------------------------------------------------------------------
__global__ __launch_bounds__(256) void k_gemm_out(const uint16_t* __restrict__ X,
                                                  const uint16_t* __restrict__ Wt,
                                                  float* __restrict__ Out) {
  const int lane = threadIdx.x & 31;
  const int wave = threadIdx.x >> 5;
  const int half = lane >> 4;
  const int l15  = lane & 15;
  const int m0 = blockIdx.x * 16;
  const int n0 = blockIdx.y * 512 + wave * 64;

  v8f acc[4] = {};
  const uint16_t* arow  = X  + (size_t)(m0 + l15) * kD;
  const uint16_t* brow0 = Wt + (size_t)(n0 + l15) * kD;

  Frags fA, fB;
  load_set(fA, arow, brow0, 0, half);
  for (int k0 = 0; k0 < kD; k0 += 64) {
    load_set(fB, arow, brow0, k0 + 32, half);
    mma_set(acc, fA);
    if (k0 + 64 < kD) load_set(fA, arow, brow0, k0 + 64, half);
    mma_set(acc, fB);
  }

#pragma unroll
  for (int t = 0; t < 4; ++t)
#pragma unroll
    for (int r = 0; r < 8; ++r) {
      int m = m0 + r + 8 * half;
      Out[(size_t)m * kD + n0 + 16 * t + l15] = acc[t][r];
    }
}

// ---------------------------------------------------------------------------
// Host-side launch
// ---------------------------------------------------------------------------
extern "C" void kernel_launch(void* const* d_in, const int* in_sizes, int n_in,
                              void* d_out, int out_size, void* d_ws, size_t ws_size,
                              hipStream_t stream) {
  (void)in_sizes; (void)n_in; (void)out_size; (void)ws_size;

  const float* x  = (const float*)d_in[0];
  const float* wq = (const float*)d_in[1];
  const float* wk = (const float*)d_in[2];
  const float* wv = (const float*)d_in[3];
  const float* wo = (const float*)d_in[4];
  float* out = (float*)d_out;

  const size_t BSD = (size_t)kB * kS * kD;   // 8,388,608 elems
  const size_t DD  = (size_t)kD * kD;        // 1,048,576 elems

  uint16_t* ws    = (uint16_t*)d_ws;
  uint16_t* xb    = ws;                      // bf16 x           [M,D]
  uint16_t* wqt   = xb  + BSD;               // bf16 wq^T        [N,K]
  uint16_t* wkt   = wqt + DD;
  uint16_t* wvt   = wkt + DD;
  uint16_t* wot   = wvt + DD;
  uint16_t* Qb    = wot + DD;                // bf16 Q           [B,H,S,DK]
  uint16_t* Kb    = Qb  + BSD;               // bf16 K           [B,H,S,DK]
  uint16_t* Vtb   = Kb  + BSD;               // bf16 V^T         [B,H,DK,S]
  uint16_t* attnb = Vtb + BSD;               // bf16 context     [B,S,D]

  // 1) conversions: x straight (vectorized), weights transposed (LDS tiles)
  k_cvt<<<2048, 256, 0, stream>>>(x, xb, (int)(BSD / 8));
  dim3 tgrid(kD / 32, kD / 32), tblk(32, 8);
  k_cvt_w_t<<<tgrid, tblk, 0, stream>>>(wq, wqt);
  k_cvt_w_t<<<tgrid, tblk, 0, stream>>>(wk, wkt);
  k_cvt_w_t<<<tgrid, tblk, 0, stream>>>(wv, wvt);
  k_cvt_w_t<<<tgrid, tblk, 0, stream>>>(wo, wot);

  // 2) Q/K/V projections (V stored transposed for contiguous P*V fragments)
  dim3 ggrid(kM / 16, kD / 512);
  k_proj<<<ggrid, 256, 0, stream>>>(xb, wqt, Qb, 0);
  k_proj<<<ggrid, 256, 0, stream>>>(xb, wkt, Kb, 0);
  k_proj<<<ggrid, 256, 0, stream>>>(xb, wvt, Vtb, 1);

  // 3) causal flash attention: 8192 query-tile waves / 4 per block
  k_attention<<<2048, 128, 0, stream>>>(Qb, Kb, Vtb, attnb);

  // 4) output projection (fp32 store)
  k_gemm_out<<<ggrid, 256, 0, stream>>>(attnb, wot, out);
}
